// OuterMean_66821101191601
// MI455X (gfx1250) — matrix operand order repaired and checked by reading, not compiled, not run on hardware
//
#include <hip/hip_runtime.h>
#include <math.h>

// Problem constants (match reference)
static constexpr int BB  = 2;    // batch
static constexpr int NN  = 512;  // sequence
static constexpr int HH  = 256;  // hidden = DIM
static constexpr float EPS = 1e-5f;

typedef __attribute__((ext_vector_type(16))) _Float16 v16h;
typedef __attribute__((ext_vector_type(8)))  _Float16 v8h;
typedef __attribute__((ext_vector_type(8)))  float    v8f;
typedef __attribute__((ext_vector_type(4)))  float    v4f;

static constexpr int    WO_STRIDE  = HH + 8;                      // halves/row (padded)
static constexpr size_t SMEM_WO    = (size_t)HH * WO_STRIDE * 2;  // 135168 B
static constexpr size_t SMEM_BYTES = SMEM_WO + 4 * HH * 4;        // + sLi,sG,sB,sBo
static constexpr size_t LEFT_BYTES = (size_t)BB * NN * HH * 4;    // 1 MB in d_ws
// WoT (padded f16) lives in d_ws right after `left`
static constexpr size_t WOT_OFFSET = LEFT_BYTES;

// ---------------------------------------------------------------------------
// Stage 0: one-time Wo -> transposed, padded f16 WoT in global scratch.
// Trivial size (132 KB); enables async bulk copy into LDS in stage 2.
// ---------------------------------------------------------------------------
__global__ __launch_bounds__(256) void om_prep_wot(
    const float* __restrict__ Wo, _Float16* __restrict__ wot)
{
    const int kk = blockIdx.x;       // 0..WO_STRIDE-1 (includes pad columns)
    const int n  = threadIdx.x;      // 0..255, coalesced read of Wo row kk
    const _Float16 v = (kk < HH) ? (_Float16)Wo[(size_t)kk * HH + n]
                                 : (_Float16)0.0f;
    wot[(size_t)n * WO_STRIDE + kk] = v;
}

// ---------------------------------------------------------------------------
// Stage 1: left[b,n,:] = LN(x[b,n,:]) @ Wl + bl, zeroed where !mask[b,n].
// One 256-thread block per row. ~0.07 GFLOP total — negligible.
// ---------------------------------------------------------------------------
__global__ __launch_bounds__(256) void om_ln_linear(
    const float* __restrict__ x, const int* __restrict__ mask,
    const float* __restrict__ ln_g, const float* __restrict__ ln_b,
    const float* __restrict__ Wl, const float* __restrict__ bl,
    float* __restrict__ left)
{
    __shared__ float sxn[HH];
    __shared__ float sred[8];
    const int row = blockIdx.x;            // b*N + n
    const int t   = threadIdx.x;

    const float xv = x[(size_t)row * HH + t];

    auto blockSum = [&](float v) -> float {
        #pragma unroll
        for (int o = 16; o > 0; o >>= 1) v += __shfl_xor(v, o, 32);
        if ((t & 31) == 0) sred[t >> 5] = v;
        __syncthreads();
        float r = 0.0f;
        #pragma unroll
        for (int w = 0; w < 8; ++w) r += sred[w];
        __syncthreads();
        return r;
    };

    const float mu = blockSum(xv) * (1.0f / HH);
    const float d  = xv - mu;
    const float var = blockSum(d * d) * (1.0f / HH);
    const float rs  = rsqrtf(var + EPS);

    sxn[t] = d * rs * ln_g[t] + ln_b[t];
    __syncthreads();

    float acc = bl[t];
    #pragma unroll 4
    for (int k = 0; k < HH; ++k)
        acc = fmaf(sxn[k], Wl[(size_t)k * HH + t], acc);   // coalesced over t

    left[(size_t)row * HH + t] = mask[row] ? acc : 0.0f;   // mask folded in here
}

// ---------------------------------------------------------------------------
// Stage 2: for each (b,i,j): out = LN(gelu(left_i (.) left_j)) @ Wo + bo.
// Block = 8 waves; wave = tile of 16 (i, j..j+15) pairs.
//  * WoT pulled into LDS with global_load_async_to_lds_b128 (ASYNCcnt),
//    fully overlapped with fragment construction.
//  * Transposed GEMM: A = WoT tile (LDS), B = h fragments (registers), so each
//    lane's 8 accumulator lanes are consecutive output columns -> b128 NT stores.
//  * Bias folded into the WMMA C operand (no post-add, no hazard NOPs).
// ---------------------------------------------------------------------------
__global__ __launch_bounds__(256) void om_outer_gemm(
    const float* __restrict__ left, const _Float16* __restrict__ wot,
    const float* __restrict__ bo,   const float* __restrict__ lno_g,
    const float* __restrict__ lno_b, float* __restrict__ out)
{
    extern __shared__ char smem[];
    _Float16* sWoT = (_Float16*)smem;                 // [n][k] padded, f16
    float* sLi = (float*)(smem + SMEM_WO);            // left_i row
    float* sG  = sLi + HH;                            // lno_g
    float* sB  = sG  + HH;                            // lno_b
    float* sBo = sB  + HH;                            // bo

    const int tid   = threadIdx.x;
    const int b     = blockIdx.z;
    const int i     = blockIdx.y;
    const int jbase = blockIdx.x * 128;

    // --- Kick off async bulk copy of WoT (global, f16, padded) into LDS.
    // 8448 x 16B transfers; each of the 8 waves issues 33 instructions.
    // Tracked by ASYNCcnt; overlapped with everything until the GEMM.
    {
        const unsigned ldsBase = (unsigned)(size_t)(void*)sWoT;
        for (int it = 0; it < 33; ++it) {
            const unsigned off = (unsigned)(it * 4096 + tid * 16);
            asm volatile(
                "global_load_async_to_lds_b128 %0, %1, %2 offset:0"
                :
                : "v"(ldsBase + off), "v"(off), "s"(wot)
                : "memory");
        }
    }

    // --- Small LDS fills (independent of WoT region) ---
    sLi[tid] = left[((size_t)b * NN + i) * HH + tid];
    sG[tid]  = lno_g[tid];
    sB[tid]  = lno_b[tid];
    sBo[tid] = bo[tid];
    __syncthreads();

    const int wave  = tid >> 5;
    const int lane  = tid & 31;
    const int m     = lane & 15;     // pair row (j offset) this lane serves
    const int hi    = lane >> 4;     // K-half selector in fragment layouts
    const int j     = jbase + wave * 16 + m;

    const float* __restrict__ lj = left + ((size_t)b * NN + j) * HH;

    // --- Pass 1: y = gelu(left_i * left_j) into h B-fragments (f16) + stats ---
    // B 32x16 f16 layout: lane(m, hi); vec elem e -> k = 32*s + 16*hi + e
    v16h hfrag[8];
    float sum = 0.0f, sumsq = 0.0f;
    #pragma unroll
    for (int s = 0; s < 8; ++s) {
        const int koff = 32 * s + 16 * hi;
        float pj[16], pi[16];
        #pragma unroll
        for (int q = 0; q < 4; ++q) {
            *(float4*)&pj[q * 4] = *(const float4*)(lj  + koff + q * 4);
            *(float4*)&pi[q * 4] = *(const float4*)(sLi + koff + q * 4);
        }
        #pragma unroll
        for (int e = 0; e < 16; ++e) {
            const float y = pi[e] * pj[e];
            const float g = 0.5f * y * (1.0f + erff(y * 0.70710678f)); // exact gelu
            sum   += g;
            sumsq += g * g;
            hfrag[s][e] = (_Float16)g;
        }
    }

    // Row stats live across lane pair (m, m+16): butterfly over bit 4.
    sum   += __shfl_xor(sum,   16, 32);
    sumsq += __shfl_xor(sumsq, 16, 32);
    const float mu  = sum * (1.0f / HH);
    const float var = sumsq * (1.0f / HH) - mu * mu;
    const float rs  = rsqrtf(var + EPS);

    // --- Pass 2: normalize fragments in place: (y-mu)*rs*g + b ---
    #pragma unroll
    for (int s = 0; s < 8; ++s) {
        const int koff = 32 * s + 16 * hi;
        #pragma unroll
        for (int e = 0; e < 16; ++e) {
            const float y = (float)hfrag[s][e];
            hfrag[s][e] = (_Float16)((y - mu) * rs * sG[koff + e] + sB[koff + e]);
        }
    }

    // --- Drain the async WoT copy; make it visible to all waves ---
    asm volatile("s_wait_asynccnt 0" ::: "memory");
    __syncthreads();

    // --- GEMM (transposed): D = WoT_tile[16x256] @ h^T  over 16 N-tiles ---
    // A 16x32 f16 layout (WoT): lane(n=lane&15, hi);
    //   elem e -> k = 32*s + 16*hi + e (same contiguous order as B)
    // D element (VGPR r, lane): out column n = t*16 + r + 8*hi, row m = lane&15
    const int n_lane = lane & 15;
    const size_t rowbase =
        (((size_t)b * NN + i) * NN + (size_t)j) * HH;   // this lane's out row

    for (int t = 0; t < 16; ++t) {
        // Seed accumulator with the output bias (C operand of the WMMA).
        const int ncol = t * 16 + 8 * hi;               // first of 8 columns
        union { v8f v; v4f q[2]; } acc;
        acc.q[0] = *(const v4f*)(sBo + ncol);
        acc.q[1] = *(const v4f*)(sBo + ncol + 4);

        #pragma unroll
        for (int s = 0; s < 8; ++s) {
            const _Float16* ap =
                &sWoT[(t * 16 + n_lane) * WO_STRIDE + 32 * s + 16 * hi];
            union { v16h v; v8h h[2]; } af;
            af.h[0] = *(const v8h*)(ap);        // k = 32s+16hi .. +7
            af.h[1] = *(const v8h*)(ap + 8);    // k = 32s+16hi+8 .. +15
            acc.v = __builtin_amdgcn_wmma_f32_16x16x32_f16(
                false, af.v, false, hfrag[s], (short)0, acc.v, false, false);
        }
        // 8 consecutive output columns per lane -> two b128 NT stores
        __builtin_nontemporal_store(acc.q[0], (v4f*)(out + rowbase + ncol));
        __builtin_nontemporal_store(acc.q[1], (v4f*)(out + rowbase + ncol + 4));
    }
}

// ---------------------------------------------------------------------------
extern "C" void kernel_launch(void* const* d_in, const int* in_sizes, int n_in,
                              void* d_out, int out_size, void* d_ws, size_t ws_size,
                              hipStream_t stream) {
    const float* x    = (const float*)d_in[0];
    const int*   mask = (const int*)  d_in[1];
    const float* ln_g = (const float*)d_in[2];
    const float* ln_b = (const float*)d_in[3];
    const float* Wl   = (const float*)d_in[4];
    const float* bl   = (const float*)d_in[5];
    const float* Wo   = (const float*)d_in[6];
    const float* bo   = (const float*)d_in[7];
    const float* lnog = (const float*)d_in[8];
    const float* lnob = (const float*)d_in[9];
    float* out  = (float*)d_out;
    float*     left = (float*)d_ws;                          // 1 MB
    _Float16*  wot  = (_Float16*)((char*)d_ws + WOT_OFFSET); // 132 KB

    (void)in_sizes; (void)n_in; (void)out_size; (void)ws_size;

    // Stage 0: Wo -> padded f16 WoT (one-time layout conversion)
    om_prep_wot<<<dim3(WO_STRIDE), 256, 0, stream>>>(Wo, wot);

    // Stage 1: masked left = LN(x) @ Wl + bl
    om_ln_linear<<<dim3(BB * NN), 256, 0, stream>>>(
        x, mask, ln_g, ln_b, Wl, bl, left);

    // Stage 2: WMMA GEMM, WoT streamed into LDS asynchronously
    hipFuncSetAttribute((const void*)om_outer_gemm,
                        hipFuncAttributeMaxDynamicSharedMemorySize,
                        (int)SMEM_BYTES);
    om_outer_gemm<<<dim3(NN / 128, NN, BB), 256, SMEM_BYTES, stream>>>(
        left, wot, bo, lnog, lnob, out);
}
